// NNModelBasic_84293028151701
// MI455X (gfx1250) — matrix operand order repaired and checked by reading, not compiled, not run on hardware
//
#include <hip/hip_runtime.h>

#define PAST 8
#define FUTURE 16
#define BATCH 16384
#define DEXO 32
#define HDIM 256
#define NMLP 7
#define ROWS 64          // batch rows per workgroup
#define ASTR 264         // padded LDS stride (elements) for activations
#define WSTR 40          // padded LDS stride for 32-wide weight chunks

typedef __attribute__((ext_vector_type(16))) __bf16 v16bf;
typedef __attribute__((ext_vector_type(8)))  float  v8f;

union FragBF { v16bf v; unsigned int u[8]; uint4 q[2]; };
union PackBF { __bf16 h[2]; unsigned int u; };

static __device__ __forceinline__ unsigned int pack2(float a, float b) {
    PackBF p; p.h[0] = (__bf16)a; p.h[1] = (__bf16)b; return p.u;
}
static __device__ __forceinline__ void load4(uint4* tmp, const __bf16* src) {
#pragma unroll
    for (int j = 0; j < 4; ++j) tmp[j] = *(const uint4*)(src + j * 8);
}
static __device__ __forceinline__ void store4(__bf16* dst, const uint4* tmp) {
#pragma unroll
    for (int j = 0; j < 4; ++j) *(uint4*)(dst + j * 8) = tmp[j];
}

// ---------------------------------------------------------------------------
// Prep: W_mlp [7][K][N] f32 -> [7][N][K] bf16
__global__ __launch_bounds__(256) void k_prep_wmlp(const float* __restrict__ Wm,
                                                   __bf16* __restrict__ Wt) {
    int idx = blockIdx.x * 256 + threadIdx.x;          // 7*65536
    int i = idx >> 16, rem = idx & 65535;
    int k = rem >> 8, n = rem & 255;
    Wt[(size_t)i * 65536 + n * 256 + k] = (__bf16)Wm[(size_t)i * 65536 + k * 256 + n];
}

// Prep: fold 9 ExoLayers into Wt[n][k], k = layer*32 + e   ([256][288] bf16)
__global__ __launch_bounds__(256) void k_prep_wexo(const float* __restrict__ We,
                                                   __bf16* __restrict__ Wt) {
    int idx = blockIdx.x * 256 + threadIdx.x;          // 288*256
    int k = idx >> 8, n = idx & 255;
    Wt[(size_t)n * 288 + k] = (__bf16)We[(size_t)k * 256 + n];
}

// Prep: W_endo [8][1][256] f32 -> [256][8] bf16 (transposed)
__global__ __launch_bounds__(256) void k_prep_wendo(const float* __restrict__ We,
                                                    __bf16* __restrict__ Wt) {
    int idx = blockIdx.x * 256 + threadIdx.x;          // 256*8
    int n = idx >> 3, k = idx & 7;
    Wt[(size_t)n * 8 + k] = (__bf16)We[(size_t)k * 256 + n];
}

// ---------------------------------------------------------------------------
// Fully fused autoregressive forecaster. One workgroup owns 64 batch rows for
// all 16 steps; activations + carry ring in LDS; weights streamed L2->LDS in
// software-pipelined 32-wide chunks (prefetch next chunk under current WMMAs).
__global__ __launch_bounds__(256) void k_fused(
    const float*  __restrict__ xs,       // [24][B][32]
    const float*  __restrict__ ys_init,  // [8][B]
    const __bf16* __restrict__ wexo_t,   // [256][288]
    const __bf16* __restrict__ wendo_t,  // [256][8]
    const float*  __restrict__ b_endo0,  // [256]
    const __bf16* __restrict__ wmlp_t,   // [7][256][256] (N-major)
    const float*  __restrict__ b_mlp,    // [7][256]
    const float*  __restrict__ W_last,   // [256]
    const float*  __restrict__ b_last,   // [1]
    float*        __restrict__ out)      // [16][B]
{
    extern __shared__ __align__(16) char smem[];
    __bf16* actA   = (__bf16*)smem;                       // [ROWS][ASTR]
    __bf16* actB   = actA + ROWS * ASTR;                  // [ROWS][ASTR]
    __bf16* sW0    = actB + ROWS * ASTR;                  // [HDIM][WSTR]
    __bf16* sW1    = sW0 + HDIM * WSTR;                   // [HDIM][WSTR]
    float*  sCarry = (float*)(sW1 + HDIM * WSTR);         // [24][ROWS]
    float*  sWlast = sCarry + (PAST + FUTURE) * ROWS;     // [256]

    const int t    = threadIdx.x;
    const int wave = t >> 5, lane = t & 31;
    const int mrow = (wave >> 1) * 16;                    // wave's 16-row band
    const int nbase = (wave & 1) * 128;                   // wave's 128-col half
    const int lh = lane >> 4, ll = lane & 15;
    const int m0 = blockIdx.x * ROWS;
    const int arow = mrow + ll;                           // local row for A frags

    // init carry ring (slots 0..7 = ys_init) + W_last
    for (int i = t; i < PAST * ROWS; i += 256)
        sCarry[i] = ys_init[(size_t)(i / ROWS) * BATCH + m0 + (i % ROWS)];
    sWlast[t] = W_last[t];
    __syncthreads();

    __bf16* actIn  = actA;
    __bf16* actOut = actB;

    for (int f = 0; f < FUTURE; ++f) {
        // ============ exo (9 pipelined chunks) + endo (1 reg chunk) ============
        v8f acc[8];
#pragma unroll
        for (int i = 0; i < 8; ++i) { v8f z = {0.f,0.f,0.f,0.f,0.f,0.f,0.f,0.f}; acc[i] = z; }

        {   // endo chunk: K=8 zero-padded to 32, branch-free masking
            const float m = (lh == 0) ? 1.f : 0.f;
            const unsigned int bm = (lh == 0) ? 0xFFFFFFFFu : 0u;
            FragBF a;
#pragma unroll
            for (int p = 0; p < 4; ++p) {
                float c0 = sCarry[(f + 2 * p)     * ROWS + arow];
                float c1 = sCarry[(f + 2 * p + 1) * ROWS + arow];
                a.u[p]     = pack2(c0 * m, c1 * m);
                a.u[p + 4] = 0u;
            }
#pragma unroll
            for (int tile = 0; tile < 8; ++tile) {
                int col = nbase + tile * 16 + ll;
                FragBF b;
#pragma unroll
                for (int p = 0; p < 4; ++p) {
                    unsigned int w = *(const unsigned int*)(wendo_t + (size_t)col * 8 + 2 * p);
                    b.u[p]     = w & bm;
                    b.u[p + 4] = 0u;
                }
                acc[tile] = __builtin_amdgcn_wmma_f32_16x16x32_bf16(
                    false, a.v, false, b.v, (short)0, acc[tile], false, false);
            }
        }

        {   // prime exo chunk 0
            uint4 tmp[4];
            load4(tmp, wexo_t + (size_t)t * 288);
            store4(sW0 + t * WSTR, tmp);
        }
        __syncthreads();

#pragma unroll
        for (int n = 0; n < PAST + 1; ++n) {
            __bf16* curW = (n & 1) ? sW1 : sW0;
            __bf16* nxtW = (n & 1) ? sW0 : sW1;
            uint4 tmp[4];
            if (n < PAST) load4(tmp, wexo_t + (size_t)t * 288 + (n + 1) * 32);

            FragBF a;   // A frag from xs window, f32 -> packed bf16 in regs
            const float* xsrow = xs + ((size_t)(f + n) * BATCH + m0 + arow) * 32;
#pragma unroll
            for (int p = 0; p < 8; ++p) {
                int kb = p * 2 + lh * 16;
                float2 v = *(const float2*)(xsrow + kb);
                a.u[p] = pack2(v.x, v.y);
            }
#pragma unroll
            for (int tile = 0; tile < 8; ++tile) {
                int col = nbase + tile * 16 + ll;
                FragBF b;
                b.q[0] = *(const uint4*)&curW[col * WSTR + lh * 16];
                b.q[1] = *(const uint4*)&curW[col * WSTR + lh * 16 + 8];
                acc[tile] = __builtin_amdgcn_wmma_f32_16x16x32_bf16(
                    false, a.v, false, b.v, (short)0, acc[tile], false, false);
            }
            if (n < PAST) {
                store4(nxtW + t * WSTR, tmp);
                __syncthreads();
            }
        }

        // epilogue: + b_endo0, ReLU -> actIn (bf16)
#pragma unroll
        for (int tile = 0; tile < 8; ++tile) {
            int col = nbase + tile * 16 + ll;
            float bv = b_endo0[col];
#pragma unroll
            for (int r = 0; r < 8; ++r) {
                int row = mrow + r + lh * 8;
                actIn[row * ASTR + col] = (__bf16)fmaxf(acc[tile][r] + bv, 0.f);
            }
        }
        __syncthreads();

        // ============ 7 MLP layers, pipelined 32-wide weight chunks ============
        for (int i = 0; i < NMLP; ++i) {
            const __bf16* wsrc = wmlp_t + (size_t)i * HDIM * HDIM + (size_t)t * HDIM;
            {   // prime chunk 0
                uint4 tmp[4];
                load4(tmp, wsrc);
                store4(sW0 + t * WSTR, tmp);
            }
            __syncthreads();

            v8f acc2[8];
#pragma unroll
            for (int q = 0; q < 8; ++q) { v8f z = {0.f,0.f,0.f,0.f,0.f,0.f,0.f,0.f}; acc2[q] = z; }

#pragma unroll
            for (int c = 0; c < 8; ++c) {
                __bf16* curW = (c & 1) ? sW1 : sW0;
                __bf16* nxtW = (c & 1) ? sW0 : sW1;
                uint4 tmp[4];
                if (c < 7) load4(tmp, wsrc + (c + 1) * 32);   // prefetch next chunk

                const int kk = c * 32;
                FragBF a;
                a.q[0] = *(const uint4*)&actIn[arow * ASTR + kk + lh * 16];
                a.q[1] = *(const uint4*)&actIn[arow * ASTR + kk + lh * 16 + 8];
#pragma unroll
                for (int tile = 0; tile < 8; ++tile) {
                    int col = nbase + tile * 16 + ll;
                    FragBF b;
                    b.q[0] = *(const uint4*)&curW[col * WSTR + lh * 16];
                    b.q[1] = *(const uint4*)&curW[col * WSTR + lh * 16 + 8];
                    acc2[tile] = __builtin_amdgcn_wmma_f32_16x16x32_bf16(
                        false, a.v, false, b.v, (short)0, acc2[tile], false, false);
                }
                if (c < 7) {
                    store4(nxtW + t * WSTR, tmp);
                    __syncthreads();
                }
            }

            const float* bias = b_mlp + (size_t)i * HDIM;
#pragma unroll
            for (int tile = 0; tile < 8; ++tile) {
                int col = nbase + tile * 16 + ll;
                float bv = bias[col];
#pragma unroll
                for (int r = 0; r < 8; ++r) {
                    int row = mrow + r + lh * 8;
                    actOut[row * ASTR + col] = (__bf16)fmaxf(acc2[tile][r] + bv, 0.f);
                }
            }
            __syncthreads();
            __bf16* tmp = actIn; actIn = actOut; actOut = tmp;
        }

        // ============ final H->1 dot, carry update, output ============
        {
            int rloc = wave * 8 + (lane >> 2);           // 8 rows per wave, 4 lanes/row
            int h0 = (lane & 3) * 64;
            const __bf16* ap = actIn + rloc * ASTR + h0;
            const float*  wl = sWlast + h0;
            float s = 0.f;
#pragma unroll
            for (int j = 0; j < 64; ++j) s += (float)ap[j] * wl[j];
            s += __shfl_xor(s, 1, 32);
            s += __shfl_xor(s, 2, 32);
            if ((lane & 3) == 0) {
                float y = s + b_last[0];
                sCarry[(PAST + f) * ROWS + rloc] = y;
                out[(size_t)f * BATCH + m0 + rloc] = y;
            }
        }
        __syncthreads();
    }
}

// ---------------------------------------------------------------------------
extern "C" void kernel_launch(void* const* d_in, const int* in_sizes, int n_in,
                              void* d_out, int out_size, void* d_ws, size_t ws_size,
                              hipStream_t stream) {
    const float* xs_all  = (const float*)d_in[0];   // [24][B][32]
    const float* ys_init = (const float*)d_in[1];   // [8][B][1]
    const float* W_exo   = (const float*)d_in[2];   // [9][32][256]
    const float* W_endo  = (const float*)d_in[3];   // [8][1][256]
    const float* b_endo0 = (const float*)d_in[4];   // [256]
    const float* W_mlp   = (const float*)d_in[5];   // [7][256][256]
    const float* b_mlp   = (const float*)d_in[6];   // [7][256]
    const float* W_last  = (const float*)d_in[7];   // [256][1]
    const float* b_last  = (const float*)d_in[8];   // [1]
    float* out = (float*)d_out;                     // [16][B][1]

    char* p = (char*)d_ws;
    __bf16* wt_mlp  = (__bf16*)p; p += (size_t)NMLP * HDIM * HDIM * 2;
    __bf16* wt_exo  = (__bf16*)p; p += (size_t)(PAST + 1) * DEXO * HDIM * 2;
    __bf16* wt_endo = (__bf16*)p; p += (size_t)HDIM * PAST * 2;

    k_prep_wmlp<<<NMLP * 256, 256, 0, stream>>>(W_mlp, wt_mlp);
    k_prep_wexo<<<(PAST + 1) * DEXO, 256, 0, stream>>>(W_exo, wt_exo);
    k_prep_wendo<<<PAST, 256, 0, stream>>>(W_endo, wt_endo);

    size_t smem = (size_t)(2 * ROWS * ASTR + 2 * HDIM * WSTR) * sizeof(__bf16)
                + (size_t)((PAST + FUTURE) * ROWS + HDIM) * sizeof(float);

    k_fused<<<BATCH / ROWS, 256, smem, stream>>>(xs_all, ys_init, wt_exo, wt_endo,
                                                 b_endo0, wt_mlp, b_mlp,
                                                 W_last, b_last, out);
}
